// DirectionalFreqEmbed_87213605912679
// MI455X (gfx1250) — matrix operand order repaired and checked by reading, not compiled, not run on hardware
//
#include <hip/hip_runtime.h>

// ---- problem constants (from reference) ----
#define IMG     64
#define CIN     30
#define DIM     384
#define BATCH   64

// ---- tiling ----
#define KB      32          // K per WMMA bf16 step
#define NB      128         // DIM columns per workgroup (3 slabs cover 384)
#define THREADS 256         // 8 waves (wave32)
#define PAD     16          // LDS row pad (bf16 elems) to spread banks
#define KBP     (KB + PAD)
#define LMAX_UB 1504        // >= roundup(Lmax=1452, 32); compile-time LDS bound

typedef __attribute__((ext_vector_type(16))) __bf16 v16bf;
typedef __attribute__((ext_vector_type(8)))  __bf16 v8bf;
typedef __attribute__((ext_vector_type(4)))  __bf16 v4bf;
typedef __attribute__((ext_vector_type(8)))  float  v8f;
typedef __attribute__((ext_vector_type(4)))  float  v4f;
typedef __attribute__((ext_vector_type(8)))  int    v8i;

__global__ __launch_bounds__(THREADS)
void dfe_wmma_kernel(const float* __restrict__ x,
                     const float* __restrict__ W,
                     const float* __restrict__ bias,
                     const int*   __restrict__ idx_a,
                     const int*   __restrict__ idx_b,
                     const int*   __restrict__ idx_c,
                     const int*   __restrict__ lens,
                     float*       __restrict__ out,
                     int T, int Lmax)
{
    // bf16 hi/lo staging planes: g (A, [b][k]) and W (B, [d][k], transposed at store)
    __shared__ __attribute__((aligned(64))) __bf16 gHi[BATCH][KBP];
    __shared__ __attribute__((aligned(64))) __bf16 gLo[BATCH][KBP];
    __shared__ __attribute__((aligned(64))) __bf16 wHi[NB][KBP];
    __shared__ __attribute__((aligned(64))) __bf16 wLo[NB][KBP];
    // precomputed gather offsets for this token
    __shared__ __attribute__((aligned(32))) int    offs[LMAX_UB];

    const int t    = blockIdx.y;
    const int n0   = blockIdx.x * NB;          // column base within DIM
    const int tid  = threadIdx.x;
    const int lane = tid & 31;
    const int wave = tid >> 5;
    const int m0   = (wave & 3) * 16;          // batch-row base of this wave
    const int ng   = (wave >> 2) * 64;         // column base of this wave within NB
    const int len  = lens[t];

    const int laneM  = lane & 15;              // row (A) / column (B,C) within 16
    const int hiHalf = lane >> 4;              // 0: lanes 0-15, 1: lanes 16-31

    // ---- one-time: precompute x-gather offsets for all l (zero beyond len) ----
    const int KR = (len + KB - 1) & ~(KB - 1);
    for (int l = tid; l < KR; l += THREADS) {
        int off = 0;
        if (l < len) {
            int ia = idx_a[t * Lmax + l];
            int ib = idx_b[t * Lmax + l];
            int ic = idx_c[t * Lmax + l];
            off = (ic * IMG + ia) * IMG + ib;
        }
        offs[l] = off;
    }
    __syncthreads();

    // staging thread mappings
    const int dq  = (tid & 31) << 2;           // W: 4-column base (0..124)
    const int kq  = (tid >> 5) << 2;           // W: 4-row base    (0..28)
    const int gb  = tid >> 2;                  // g: batch row     (0..63)
    const int gko = (tid & 3) << 3;            // g: 8-k base      (0,8,16,24)
    const float* Wt = W + (long)t * Lmax * DIM + n0 + dq;
    const float* xb = x + gb * (CIN * IMG * IMG);

    v8f acc[4] = {};                           // 4 x (16x16 f32) C tiles

    for (int k0 = 0; k0 < len; k0 += KB) {
        const bool tail = (k0 + KB > len);     // uniform across block

        // prefetch next W tile rows into cache while we stage this one
        if (k0 + KB < len)
            __builtin_prefetch(Wt + (long)(k0 + KB + kq) * DIM, 0, 1);

        // ---- stage W: each thread owns a 4(k) x 4(d) block; packed b64 stores ----
        v4f r[4];
        #pragma unroll
        for (int rr = 0; rr < 4; ++rr) {
            int krow = k0 + kq + rr;
            if (tail) krow = min(krow, len - 1);   // in-bounds garbage; g==0 masks it
            r[rr] = *(const v4f*)(Wt + (long)krow * DIM);
        }
        #pragma unroll
        for (int j = 0; j < 4; ++j) {
            v4bf h, lo;
            #pragma unroll
            for (int rr = 0; rr < 4; ++rr) {
                float  f  = r[rr][j];
                __bf16 hb = (__bf16)f;
                h[rr]  = hb;
                lo[rr] = (__bf16)(f - (float)hb);
            }
            *(v4bf*)(&wHi[dq + j][kq]) = h;        // ds_store_b64
            *(v4bf*)(&wLo[dq + j][kq]) = lo;
        }

        // ---- stage g: each thread owns 8 consecutive k for one batch row ----
        {
            v8i  o = *(const v8i*)(&offs[k0 + gko]);   // 32B LDS read
            v8bf h, lo;
            #pragma unroll
            for (int i = 0; i < 8; ++i) {
                float f = xb[o[i]];                    // unconditional (off in-bounds)
                if (tail && (k0 + gko + i >= len)) f = 0.0f;  // cndmask in tail only
                __bf16 hb = (__bf16)f;
                h[i]  = hb;
                lo[i] = (__bf16)(f - (float)hb);
            }
            *(v8bf*)(&gHi[gb][gko]) = h;               // ds_store_b128
            *(v8bf*)(&gLo[gb][gko]) = lo;
        }
        __syncthreads();

        // ---- A fragments (16x32 bf16): lanes 0-15 K=0..15, lanes 16-31 K=16..31
        const int akb = hiHalf ? 16 : 0;
        v16bf aHi, aLo;
        {
            const __bf16* pH = &gHi[m0 + laneM][0];
            const __bf16* pL = &gLo[m0 + laneM][0];
            v8bf h0 = *(const v8bf*)(pH + akb);
            v8bf h1 = *(const v8bf*)(pH + akb + 8);
            v8bf l0 = *(const v8bf*)(pL + akb);
            v8bf l1 = *(const v8bf*)(pL + akb + 8);
            #pragma unroll
            for (int j = 0; j < 8; ++j) {
                aHi[j] = h0[j]; aHi[j + 8] = h1[j];
                aLo[j] = l0[j]; aLo[j + 8] = l1[j];
            }
        }
        // ---- B fragments (32x16 bf16): lanes 0-15 K=0..15, lanes 16-31 K=16..31, N=lane&15
        const int bkb = hiHalf ? 16 : 0;
        #pragma unroll
        for (int nt = 0; nt < 4; ++nt) {
            int col = ng + nt * 16 + laneM;
            v16bf bHi = *(const v16bf*)(&wHi[col][bkb]);
            v16bf bLo = *(const v16bf*)(&wLo[col][bkb]);
            // split-precision product: hi*hi + hi*lo + lo*hi, f32 accumulate
            acc[nt] = __builtin_amdgcn_wmma_f32_16x16x32_bf16(
                false, aHi, false, bHi, (short)0, acc[nt], false, false);
            acc[nt] = __builtin_amdgcn_wmma_f32_16x16x32_bf16(
                false, aHi, false, bLo, (short)0, acc[nt], false, false);
            acc[nt] = __builtin_amdgcn_wmma_f32_16x16x32_bf16(
                false, aLo, false, bHi, (short)0, acc[nt], false, false);
        }
        __syncthreads();
    }

    // ---- epilogue: C layout VGPR r -> (M = m0 + 8*hiHalf + r, N = laneM); add bias, store
    const int bRow = m0 + hiHalf * 8;
    #pragma unroll
    for (int nt = 0; nt < 4; ++nt) {
        int   d  = n0 + ng + nt * 16 + laneM;
        float bv = bias[t * DIM + d];
        #pragma unroll
        for (int r = 0; r < 8; ++r) {
            int b = bRow + r;
            out[((long)b * T + t) * DIM + d] = acc[nt][r] + bv;
        }
    }
}

extern "C" void kernel_launch(void* const* d_in, const int* in_sizes, int n_in,
                              void* d_out, int out_size, void* d_ws, size_t ws_size,
                              hipStream_t stream)
{
    const float* x     = (const float*)d_in[0];
    const float* W     = (const float*)d_in[1];
    const float* bias  = (const float*)d_in[2];
    const int*   idx_a = (const int*)d_in[3];
    const int*   idx_b = (const int*)d_in[4];
    const int*   idx_c = (const int*)d_in[5];
    const int*   lens  = (const int*)d_in[6];
    float*       out   = (float*)d_out;

    const int T    = in_sizes[6];            // 240
    const int Lmax = in_sizes[3] / T;        // ~1452

    dim3 grid(DIM / NB, T);                  // 3 x 240 workgroups
    dim3 block(THREADS);
    dfe_wmma_kernel<<<grid, block, 0, stream>>>(x, W, bias, idx_a, idx_b, idx_c,
                                                lens, out, T, Lmax);
    (void)d_ws; (void)ws_size; (void)n_in; (void)out_size;
}
